// SwinTransformerBlock_83837761618196
// MI455X (gfx1250) — matrix operand order repaired and checked by reading, not compiled, not run on hardware
//
#include <hip/hip_runtime.h>
#include <math.h>

// ---------------------------------------------------------------------------
// Swin block, MI455X (gfx1250, wave32). All GEMMs via V_WMMA_F32_16X16X32_BF16.
// Window staging via TDM tensor_load_to_lds (TENSORcnt).
// ---------------------------------------------------------------------------

typedef __attribute__((ext_vector_type(16))) __bf16 v16bf;
typedef __attribute__((ext_vector_type(8)))  float  v8f;
typedef unsigned int u32x4 __attribute__((ext_vector_type(4)));
typedef unsigned int u32x8 __attribute__((ext_vector_type(8)));

union Frag { unsigned int u[8]; v16bf v; };

__device__ __forceinline__ unsigned short f2bf(float f) {
  unsigned int u = __float_as_uint(f);
  u += 0x7FFFu + ((u >> 16) & 1u);          // round-to-nearest-even
  return (unsigned short)(u >> 16);
}

__device__ __forceinline__ v8f wmma_bf16(v16bf a, v16bf b, v8f c) {
  return __builtin_amdgcn_wmma_f32_16x16x32_bf16(false, a, false, b,
                                                 (short)0, c, false, false);
}

// A fragment: row-major bf16 src, stride S elems, tile origin (row0,k0),
// rows clamped to [0,maxRow). Contiguous k order: k = 16*h + s.
__device__ __forceinline__ v16bf load_a(const unsigned short* src, int S,
                                        int row0, int k0, int maxRow) {
  int lane = threadIdx.x & 31;
  int m = lane & 15, h = lane >> 4;
  int row = row0 + m; row = row < maxRow ? row : maxRow - 1;
  const unsigned short* p = src + row * S + k0 + h * 16;
  Frag f;
#pragma unroll
  for (int j = 0; j < 4; ++j) {
    f.u[j]     = *(const unsigned int*)(p + 2 * j);
    f.u[4 + j] = *(const unsigned int*)(p + 8 + 2 * j);
  }
  return f.v;
}

// B fragment from pair-packed weights: wp layout [K/2][N] uint, tile (k0,n0).
__device__ __forceinline__ v16bf load_b_pack(const unsigned int* wp, int N,
                                             int k0, int n0) {
  int lane = threadIdx.x & 31;
  int n = n0 + (lane & 15), h = lane >> 4;
  const unsigned int* p = wp + (size_t)(k0 / 2 + h * 8) * N + n;
  Frag f;
#pragma unroll
  for (int v = 0; v < 8; ++v) f.u[v] = p[v * N];
  return f.v;
}

// B fragment where B[k][n] = src[n][k] (src row-major, GEMM-K contiguous),
// i.e. K^T for Q.K^T. Columns clamped to [0,maxN).
__device__ __forceinline__ v16bf load_b_t(const unsigned short* src, int S,
                                          int k0, int n0, int maxN) {
  int lane = threadIdx.x & 31;
  int n = n0 + (lane & 15); n = n < maxN ? n : maxN - 1;
  int h = lane >> 4;
  const unsigned short* p = src + n * S + k0 + h * 16;
  Frag f;
#pragma unroll
  for (int v = 0; v < 8; ++v) f.u[v] = *(const unsigned int*)(p + 2 * v);
  return f.v;
}

// TDM: async DMA of `dwords` contiguous dwords from global -> LDS (1-D D#).
// D# group0/group1 bit layout per CDNA5 ISA 8.3/8.4. Issue once per wave.
__device__ __forceinline__ void tdm_load_1d(unsigned lds_off, const void* gptr,
                                            unsigned dwords) {
  unsigned long long ga = (unsigned long long)(size_t)gptr;
  u32x4 g0;
  g0[0] = 1u;                                            // count=1, user-mode
  g0[1] = lds_off;                                       // lds_addr (bytes)
  g0[2] = (unsigned)ga;                                  // global_addr[31:0]
  g0[3] = (unsigned)((ga >> 32) & 0x01FFFFFFu) | (2u << 30); // addr[56:32], type=2
  u32x8 g1;
  g1[0] = 2u << 16;                                      // data_size = 4 bytes
  g1[1] = (dwords & 0xFFFFu) << 16;                      // tensor_dim0[15:0]
  g1[2] = (dwords >> 16) | (1u << 16);                   // tensor_dim0[31:16], tensor_dim1=1
  g1[3] = (dwords & 0xFFFFu) << 16;                      // tile_dim0
  g1[4] = 1u;                                            // tile_dim1=1, tile_dim2=0
  g1[5] = dwords;                                        // tensor_dim0_stride[31:0]
  g1[6] = 0u;                                            // stride hi / dim1_stride lo
  g1[7] = 0u;
  asm volatile("tensor_load_to_lds %0, %1" :: "s"(g0), "s"(g1) : "memory");
}

// ---------------------------------------------------------------------------
// K1: pack fp32 weight [K][N] -> bf16 pair-packed [K/2][N] uint
// ---------------------------------------------------------------------------
__global__ void k_pack(const float* __restrict__ src, unsigned int* __restrict__ dst,
                       int K2, int N) {
  int idx = blockIdx.x * 256 + threadIdx.x;
  if (idx >= K2 * N) return;
  int kp = idx / N, n = idx % N;
  unsigned int lo = f2bf(src[(size_t)(2 * kp) * N + n]);
  unsigned int hi = f2bf(src[(size_t)(2 * kp + 1) * N + n]);
  dst[idx] = lo | (hi << 16);
}

// ---------------------------------------------------------------------------
// K1b: bm[wpos][head][n][m] = rel_bias_table[rpi[n][m]][head] + mask[wpos][n][m]
// ---------------------------------------------------------------------------
__global__ void k_biasmask(const float* __restrict__ table, const int* __restrict__ rpi,
                           const float* __restrict__ mask, float* __restrict__ bm) {
  int idx = blockIdx.x * 256 + threadIdx.x;
  if (idx >= 64 * 6 * 49 * 49) return;
  int mcol = idx % 49; int tmp = idx / 49;
  int nrow = tmp % 49; tmp /= 49;
  int head = tmp % 6;  int wpos = tmp / 6;
  bm[idx] = table[rpi[nrow * 49 + mcol] * 6 + head] + mask[wpos * 2401 + nrow * 49 + mcol];
}

// ---------------------------------------------------------------------------
// K2: LN1 + cyclic shift + window partition -> xw bf16 [200704][192]
// 64 rows/block, 4 threads per row.
// ---------------------------------------------------------------------------
__global__ void __launch_bounds__(256)
k_ln1_window(const float* __restrict__ x, const float* __restrict__ g,
             const float* __restrict__ b, unsigned short* __restrict__ xw) {
  __shared__ float red[256], red2[256];
  __shared__ float mean_s[64], rstd_s[64];
  int t = threadIdx.x;
  int rloc = t >> 2, qq = t & 3;
  int m = blockIdx.x * 64 + rloc;                 // flat window-row 0..200703
  int w = m / 49, r = m % 49;
  int bidx = w >> 6, wpos = w & 63;
  int wh = wpos >> 3, ww = wpos & 7;
  int i = r / 7, j = r % 7;
  int hh = wh * 7 + i + 3; if (hh >= 56) hh -= 56;
  int wc = ww * 7 + j + 3; if (wc >= 56) wc -= 56;
  const float* src = x + ((size_t)bidx * 3136 + hh * 56 + wc) * 192;
  float s = 0.f, s2 = 0.f;
#pragma unroll 4
  for (int c = 0; c < 48; ++c) { float v = src[qq * 48 + c]; s += v; s2 += v * v; }
  red[t] = s; red2[t] = s2;
  __syncthreads();
  if (qq == 0) {
    float a  = red[t] + red[t + 1] + red[t + 2] + red[t + 3];
    float a2 = red2[t] + red2[t + 1] + red2[t + 2] + red2[t + 3];
    float mu = a * (1.0f / 192.0f);
    float var = a2 * (1.0f / 192.0f) - mu * mu;
    mean_s[rloc] = mu;
    rstd_s[rloc] = rsqrtf(var + 1e-5f);
  }
  __syncthreads();
  float mu = mean_s[rloc], rs = rstd_s[rloc];
  unsigned short* dst = xw + (size_t)m * 192;
#pragma unroll 4
  for (int c = 0; c < 48; ++c) {
    int col = qq * 48 + c;
    dst[col] = f2bf((src[col] - mu) * rs * g[col] + b[col]);
  }
}

// ---------------------------------------------------------------------------
// K3: fused per-window attention: TDM window stage -> QKV GEMM ->
//     softmax(QK^T+bias+mask) -> PV -> proj -> window-reverse scatter +
//     residual.  One block per window, 8 waves.
// ---------------------------------------------------------------------------
__global__ void __launch_bounds__(256)
k_attn(const unsigned short* __restrict__ xw,
       const unsigned int* __restrict__ qkv_wp, const float* __restrict__ qkv_b,
       const unsigned int* __restrict__ proj_wp, const float* __restrict__ proj_b,
       const float* __restrict__ bm, const float* __restrict__ x,
       float* __restrict__ out) {
  __shared__ __attribute__((aligned(16))) unsigned short xw_s[49 * 192];
  __shared__ __attribute__((aligned(16))) unsigned short q_s[6 * 49 * 32];
  __shared__ __attribute__((aligned(16))) unsigned short k_s[6 * 49 * 32];
  __shared__ __attribute__((aligned(16))) unsigned int   vp_s[6 * 32 * 32]; // pair-packed V
  __shared__ __attribute__((aligned(16))) float          sc_s[8 * 16 * 64]; // per-wave score strip
  __shared__ __attribute__((aligned(16))) unsigned short p_s[8 * 16 * 64];  // per-wave prob strip
  __shared__ __attribute__((aligned(16))) unsigned short o_s[49 * 192];

  const float SCALE = 0.17677669529663687f;     // 32^-0.5
  int w = blockIdx.x;
  int wpos = w & 63;
  int t = threadIdx.x;
  int wave = t >> 5, lane = t & 31, n = lane & 15, h = lane >> 4;

  // ---- stage 1: TDM async copy of this window into LDS (wave 0 issues),
  //      all waves zero the pair-packed V pad meanwhile ----
  if (wave == 0) {
    tdm_load_1d((unsigned)(size_t)(void*)xw_s,
                xw + (size_t)w * 49 * 192, 49 * 192 / 2);
    __builtin_prefetch(qkv_wp + lane * 512, 0, 0);   // warm L2 for packed weights
    __builtin_prefetch(proj_wp + lane * 512, 0, 0);
  }
  for (int i = t; i < 6 * 32 * 32; i += 256) vp_s[i] = 0;
  if (wave == 0) __builtin_amdgcn_s_wait_tensorcnt(0);
  __syncthreads();

  // ---- stage 2: QKV GEMM [49x192]@[192x576], scatter into q/k/vp ----
  for (int tile = wave; tile < 144; tile += 8) {
    int mt = tile & 3, nt = tile >> 2;
    v8f c = {};
#pragma unroll
    for (int ks = 0; ks < 6; ++ks)
      c = wmma_bf16(load_a(xw_s, 192, mt * 16, ks * 32, 49),
                    load_b_pack(qkv_wp, 576, ks * 32, nt * 16), c);
    int col = nt * 16 + n;
    int seg = col / 192, cc = col % 192, head = cc >> 5, d = cc & 31;
    float bias = qkv_b[col];
#pragma unroll
    for (int v = 0; v < 8; ++v) {
      int row = mt * 16 + v + 8 * h;
      if (row < 49) {
        float val = c[v] + bias;
        if (seg == 0)      q_s[(head * 49 + row) * 32 + d] = f2bf(val * SCALE);
        else if (seg == 1) k_s[(head * 49 + row) * 32 + d] = f2bf(val);
        else ((unsigned short*)vp_s)[head * 2048 + (row >> 1) * 64 + d * 2 + (row & 1)] = f2bf(val);
      }
    }
  }
  __syncthreads();

  // ---- stage 3: attention per (head, 16-query strip) ----
  float*          strip  = sc_s + wave * 16 * 64;
  unsigned short* pstrip = p_s  + wave * 16 * 64;
  for (int u = wave; u < 24; u += 8) {
    int head = u >> 2, mt = u & 3;
    const unsigned short* qh = q_s + head * 49 * 32;
    const unsigned short* kh = k_s + head * 49 * 32;
    const float* bmh = bm + ((size_t)wpos * 6 + head) * 49 * 49;
    // scores = q @ k^T  (+ bias + mask); pad -> -1e30
#pragma unroll
    for (int nt = 0; nt < 4; ++nt) {
      v8f c = {};
      c = wmma_bf16(load_a(qh, 32, mt * 16, 0, 49),
                    load_b_t(kh, 32, 0, nt * 16, 49), c);
      int col = nt * 16 + n;
#pragma unroll
      for (int v = 0; v < 8; ++v) {
        int row = mt * 16 + v + 8 * h;
        float s = -1e30f;
        if (row < 49 && col < 49) s = c[v] + bmh[row * 49 + col];
        strip[(v + 8 * h) * 64 + col] = s;
      }
    }
    // softmax: 2 lanes per row (cols h*32..h*32+31), combine across half-waves
    {
      int row = lane & 15;
      float mx = -1e30f;
#pragma unroll 8
      for (int j = 0; j < 32; ++j) mx = fmaxf(mx, strip[row * 64 + h * 32 + j]);
      mx = fmaxf(mx, __shfl_xor(mx, 16, 32));
      float sum = 0.f;
#pragma unroll 8
      for (int j = 0; j < 32; ++j) sum += expf(strip[row * 64 + h * 32 + j] - mx);
      sum += __shfl_xor(sum, 16, 32);
      float inv = 1.0f / sum;
#pragma unroll 8
      for (int j = 0; j < 32; ++j)
        pstrip[row * 64 + h * 32 + j] = f2bf(expf(strip[row * 64 + h * 32 + j] - mx) * inv);
    }
    // out = P @ V  (K runs over 64 padded keys; pad probs are exactly 0)
#pragma unroll
    for (int nt2 = 0; nt2 < 2; ++nt2) {
      v8f c = {};
#pragma unroll
      for (int kc = 0; kc < 2; ++kc) {
        Frag fb;
        const unsigned int* p = vp_s + head * 1024 + (kc * 16 + h * 8) * 32 + nt2 * 16 + n;
#pragma unroll
        for (int v = 0; v < 8; ++v) fb.u[v] = p[v * 32];
        c = wmma_bf16(load_a(pstrip, 64, 0, kc * 32, 16), fb.v, c);
      }
#pragma unroll
      for (int v = 0; v < 8; ++v) {
        int row = mt * 16 + v + 8 * h;
        if (row < 49) o_s[row * 192 + head * 32 + nt2 * 16 + n] = f2bf(c[v]);
      }
    }
  }
  __syncthreads();

  // ---- stage 4: proj [49x192]@[192x192] + window-reverse scatter + residual ----
  int bidx = w >> 6, wh = wpos >> 3, ww = wpos & 7;
  for (int tile = wave; tile < 48; tile += 8) {
    int mt = tile & 3, nt = tile >> 2;
    v8f c = {};
#pragma unroll
    for (int ks = 0; ks < 6; ++ks)
      c = wmma_bf16(load_a(o_s, 192, mt * 16, ks * 32, 49),
                    load_b_pack(proj_wp, 192, ks * 32, nt * 16), c);
    int col = nt * 16 + n;
    float pb = proj_b[col];
#pragma unroll
    for (int v = 0; v < 8; ++v) {
      int row = mt * 16 + v + 8 * h;
      if (row < 49) {
        int i = row / 7, j = row % 7;
        int hh = wh * 7 + i + 3; if (hh >= 56) hh -= 56;
        int wc = ww * 7 + j + 3; if (wc >= 56) wc -= 56;
        size_t idx = ((size_t)bidx * 3136 + hh * 56 + wc) * 192 + col;
        out[idx] = x[idx] + c[v] + pb;          // x' = shortcut + attn branch
      }
    }
  }
}

// ---------------------------------------------------------------------------
// K4: fused MLP, in place on d_out (holds x'): LN2 -> W1+GELU -> W2 + residual.
// 64 rows/block.
// ---------------------------------------------------------------------------
__global__ void __launch_bounds__(256)
k_mlp(float* __restrict__ xo, const float* __restrict__ g, const float* __restrict__ bb,
      const unsigned int* __restrict__ w1p, const float* __restrict__ b1,
      const unsigned int* __restrict__ w2p, const float* __restrict__ b2) {
  __shared__ __attribute__((aligned(16))) unsigned short h1[64 * 192];
  __shared__ __attribute__((aligned(16))) unsigned short hid[64 * 768];
  __shared__ float red[256], red2[256], mean_s[64], rstd_s[64];
  int t = threadIdx.x;
  size_t base = (size_t)blockIdx.x * 64 * 192;
  int rloc = t >> 2, qq = t & 3;
  if (t < 32) {
    __builtin_prefetch(w1p + t * 512, 0, 0);
    __builtin_prefetch(w2p + t * 512, 0, 0);
  }
  const float* src = xo + base + (size_t)rloc * 192;
  float s = 0.f, s2 = 0.f;
#pragma unroll 4
  for (int c = 0; c < 48; ++c) { float v = src[qq * 48 + c]; s += v; s2 += v * v; }
  red[t] = s; red2[t] = s2;
  __syncthreads();
  if (qq == 0) {
    float a  = red[t] + red[t + 1] + red[t + 2] + red[t + 3];
    float a2 = red2[t] + red2[t + 1] + red2[t + 2] + red2[t + 3];
    float mu = a * (1.0f / 192.0f);
    float var = a2 * (1.0f / 192.0f) - mu * mu;
    mean_s[rloc] = mu;
    rstd_s[rloc] = rsqrtf(var + 1e-5f);
  }
  __syncthreads();
  {
    float mu = mean_s[rloc], rs = rstd_s[rloc];
#pragma unroll 4
    for (int c = 0; c < 48; ++c) {
      int col = qq * 48 + c;
      h1[rloc * 192 + col] = f2bf((src[col] - mu) * rs * g[col] + bb[col]);
    }
  }
  __syncthreads();

  int wave = t >> 5, lane = t & 31, n = lane & 15, h = lane >> 4;
  // GEMM1: [64x192]@[192x768] + bias + exact GELU -> hid (bf16)
  for (int tile = wave; tile < 192; tile += 8) {
    int mt = tile & 3, nt = tile >> 2;
    v8f c = {};
#pragma unroll
    for (int ks = 0; ks < 6; ++ks)
      c = wmma_bf16(load_a(h1, 192, mt * 16, ks * 32, 64),
                    load_b_pack(w1p, 768, ks * 32, nt * 16), c);
    int col = nt * 16 + n;
    float bias = b1[col];
#pragma unroll
    for (int v = 0; v < 8; ++v) {
      int row = mt * 16 + v + 8 * h;
      float z = c[v] + bias;
      float gl = 0.5f * z * (1.0f + erff(z * 0.70710678118654752f));
      hid[row * 768 + col] = f2bf(gl);
    }
  }
  __syncthreads();
  // GEMM2: [64x768]@[768x192] + bias + residual, in place
  for (int tile = wave; tile < 48; tile += 8) {
    int mt = tile & 3, nt = tile >> 2;
    v8f c = {};
#pragma unroll
    for (int ks = 0; ks < 24; ++ks)
      c = wmma_bf16(load_a(hid, 768, mt * 16, ks * 32, 64),
                    load_b_pack(w2p, 192, ks * 32, nt * 16), c);
    int col = nt * 16 + n;
    float bias = b2[col];
#pragma unroll
    for (int v = 0; v < 8; ++v) {
      int row = mt * 16 + v + 8 * h;
      size_t idx = base + (size_t)row * 192 + col;
      xo[idx] = xo[idx] + c[v] + bias;
    }
  }
}

// ---------------------------------------------------------------------------
// Host launcher
// ---------------------------------------------------------------------------
extern "C" void kernel_launch(void* const* d_in, const int* in_sizes, int n_in,
                              void* d_out, int out_size, void* d_ws, size_t ws_size,
                              hipStream_t stream) {
  const float* x        = (const float*)d_in[0];
  const float* mask     = (const float*)d_in[1];
  const int*   rpi      = (const int*)  d_in[2];
  const float* ln1_g    = (const float*)d_in[3];
  const float* ln1_b    = (const float*)d_in[4];
  const float* qkv_w    = (const float*)d_in[5];
  const float* qkv_b    = (const float*)d_in[6];
  const float* proj_w   = (const float*)d_in[7];
  const float* proj_b   = (const float*)d_in[8];
  const float* rbt      = (const float*)d_in[9];
  const float* ln2_g    = (const float*)d_in[10];
  const float* ln2_b    = (const float*)d_in[11];
  const float* mlp_w1   = (const float*)d_in[12];
  const float* mlp_b1   = (const float*)d_in[13];
  const float* mlp_w2   = (const float*)d_in[14];
  const float* mlp_b2   = (const float*)d_in[15];
  float* out = (float*)d_out;

  // workspace carve (256B aligned)
  char* ws = (char*)d_ws;
  size_t off = 0;
  auto carve = [&](size_t bytes) { void* p = ws + off; off = (off + bytes + 255) & ~(size_t)255; return p; };
  unsigned short* xw      = (unsigned short*)carve((size_t)200704 * 192 * 2);
  float*          bmw     = (float*)         carve((size_t)64 * 6 * 49 * 49 * 4);
  unsigned int*   qkv_wp  = (unsigned int*)  carve((size_t)96 * 576 * 4);
  unsigned int*   proj_wp = (unsigned int*)  carve((size_t)96 * 192 * 4);
  unsigned int*   w1p     = (unsigned int*)  carve((size_t)96 * 768 * 4);
  unsigned int*   w2p     = (unsigned int*)  carve((size_t)384 * 192 * 4);
  (void)ws_size; (void)in_sizes; (void)n_in; (void)out_size;

  // K1: weight packing
  k_pack<<<(96 * 576 + 255) / 256, 256, 0, stream>>>(qkv_w,  qkv_wp,  96, 576);
  k_pack<<<(96 * 192 + 255) / 256, 256, 0, stream>>>(proj_w, proj_wp, 96, 192);
  k_pack<<<(96 * 768 + 255) / 256, 256, 0, stream>>>(mlp_w1, w1p,     96, 768);
  k_pack<<<(384 * 192 + 255) / 256, 256, 0, stream>>>(mlp_w2, w2p,    384, 192);
  // K1b: bias + mask table
  k_biasmask<<<(64 * 6 * 49 * 49 + 255) / 256, 256, 0, stream>>>(rbt, rpi, mask, bmw);
  // K2: LN1 + shift + window partition -> bf16
  k_ln1_window<<<3136, 256, 0, stream>>>(x, ln1_g, ln1_b, xw);
  // K3: fused windowed attention (+ proj, residual); writes x' to d_out
  k_attn<<<4096, 256, 0, stream>>>(xw, qkv_wp, qkv_b, proj_wp, proj_b, bmw, x, out);
  // K4: fused MLP in place on d_out
  k_mlp<<<3136, 256, 0, stream>>>(out, ln2_g, ln2_b, w1p, mlp_b1, w2p, mlp_b2);
}
